// GGNN_51539607552905
// MI455X (gfx1250) — matrix operand order repaired and checked by reading, not compiled, not run on hardware
//
#include <hip/hip_runtime.h>
#include <hip/hip_bf16.h>

typedef __bf16 bf16_t;
typedef __attribute__((ext_vector_type(16))) __bf16 v16bf;
typedef __attribute__((ext_vector_type(8)))  __bf16 v8bf;
typedef __attribute__((ext_vector_type(4)))  __bf16 v4bf;
typedef __attribute__((ext_vector_type(8)))  float   v8f;
typedef __attribute__((ext_vector_type(4)))  float   v4f;
typedef __attribute__((ext_vector_type(4)))  unsigned int u32x4;
typedef __attribute__((ext_vector_type(8)))  int          i32x8;
typedef __attribute__((ext_vector_type(4)))  int          i32x4;

#define WMMA_BF16(a,b,c) __builtin_amdgcn_wmma_f32_16x16x32_bf16(false,(a),false,(b),(short)0,(c),false,false)

constexpr int B_    = 4;
constexpr int N_    = 2048;
constexpr int E_    = 2;
constexpr int D_    = 128;
constexpr int NE    = N_ * E_;     // 4096
constexpr int TD    = 3 * D_;      // 384
constexpr int M_TOT = B_ * N_;     // 8192
constexpr int NSTEP = 5;
constexpr int KT    = 256;         // K macro-tile for the TDM path
constexpr int LDP   = 264;         // 256 + 8 elem pad -> 528B row stride

// ---------------------------------------------------------------------------
// fragment helpers (layouts per CDNA5 ISA 7.12.2, wave32)
// ---------------------------------------------------------------------------
__device__ __forceinline__ v16bf cat16(v8bf lo, v8bf hi) {
  return __builtin_shufflevector(lo, hi, 0,1,2,3,4,5,6,7,8,9,10,11,12,13,14,15);
}

// A operand: row-major bf16 (16 x 32 tile).  lane l: row=l%16, klo=(l/16)*8,
// elements {klo..klo+7, klo+16..klo+23} -> two contiguous 16B loads.
__device__ __forceinline__ v16bf load_a_rm(const bf16_t* base, int ldk, int lane) {
  int row = lane & 15;
  int klo = (lane >> 4) * 8;
  const bf16_t* p = base + (size_t)row * ldk + klo;
  v8bf lo = *(const v8bf*)(p);
  v8bf hi = *(const v8bf*)(p + 16);
  return cat16(lo, hi);
}

// B operand from N-major storage Bt[col][k]. lane l: col=l%16, kb=(l/16)*8,
// elements {kb..kb+7, kb+16..kb+23} -> two contiguous 16B loads.
__device__ __forceinline__ v16bf load_b_t(const bf16_t* base, int ldk, int lane) {
  int col = lane & 15;
  int kb  = (lane >> 4) * 8;
  const bf16_t* p = base + (size_t)col * ldk + kb;
  v8bf lo = *(const v8bf*)(p);
  v8bf hi = *(const v8bf*)(p + 16);
  return cat16(lo, hi);
}

// ---------------------------------------------------------------------------
// TDM: build D# (ISA 8.3/8.4) and issue tensor_load_to_lds of a
// 32-row x 256-elem bf16 tile, row stride 8192 elems, LDS pad 16B per 512B row.
// ---------------------------------------------------------------------------
__device__ __forceinline__ void tdm_load_tile(const void* gsrc, unsigned lds_addr) {
  unsigned long long ga = (unsigned long long)gsrc;
  u32x4 g0 = { 1u,                                   // count=1, user mode
               lds_addr,                             // lds_addr (bytes)
               (unsigned)ga,                         // global_addr[31:0]
               (unsigned)((ga >> 32) & 0x1FFFFFFu) | (2u << 30) };  // addr[56:32], type=2
  i32x8 g1 = { (int)((1u << 16) |                    // data_size = 2 bytes
                     (1u << 20) |                    // pad_enable
                     (6u << 22) |                    // pad_interval: 128 DW = 512B
                     (3u << 25)),                    // pad_amount: 4 DW = 16B
               0,                                    // barrier addr 0, tensor_dim0 lo16=0
               0x10,                                 // tensor_dim0 = 1<<20 (no clip)
               (int)(0x10u | ((unsigned)KT << 16)),  // tensor_dim1 = 1<<20; tile_dim0 = 256
               32,                                   // tile_dim1 = 32 rows, tile_dim2 = 0
               2 * NE,                               // tensor_dim0_stride = 8192 elems
               0, 0 };
  i32x4 g2 = {0, 0, 0, 0};
  i32x4 g3 = {0, 0, 0, 0};
#if __clang_major__ >= 23
  i32x8 g4 = {0, 0, 0, 0, 0, 0, 0, 0};
  __builtin_amdgcn_tensor_load_to_lds(g0, g1, g2, g3, g4, 0);
#else
  __builtin_amdgcn_tensor_load_to_lds(g0, g1, g2, g3, 0);
#endif
}

// flat->LDS offset: LDS aperture maps addr[31:0] to the LDS offset (ISA 10.2)
__device__ __forceinline__ unsigned lds_off(const void* p) {
  return (unsigned)(unsigned long long)p;
}

// ---------------------------------------------------------------------------
// K0: A f32 -> bf16 (lossless: entries are 0/1), 8 elems/thread
// ---------------------------------------------------------------------------
__global__ void k_conv_A(const float* __restrict__ src, bf16_t* __restrict__ dst) {
  size_t i = ((size_t)blockIdx.x * blockDim.x + threadIdx.x) * 8;
  v4f a = *(const v4f*)(src + i);
  v4f b = *(const v4f*)(src + i + 4);
  v8bf o;
  o[0] = (bf16_t)a[0]; o[1] = (bf16_t)a[1]; o[2] = (bf16_t)a[2]; o[3] = (bf16_t)a[3];
  o[4] = (bf16_t)b[0]; o[5] = (bf16_t)b[1]; o[6] = (bf16_t)b[2]; o[7] = (bf16_t)b[3];
  *(v8bf*)(dst + i) = o;
}

// ---------------------------------------------------------------------------
// K1: transpose-convert f32 weights (K x Nc) -> bf16 Nc-major (Nc x K)
// ---------------------------------------------------------------------------
__global__ void k_transpose_bf16(const float* __restrict__ src, bf16_t* __restrict__ dst,
                                 int K, int Nc) {
  int i = blockIdx.x * blockDim.x + threadIdx.x;
  if (i < K * Nc) {
    int k = i / Nc, f = i - k * Nc;
    dst[(size_t)f * K + k] = (bf16_t)src[i];
  }
}

// ---------------------------------------------------------------------------
// K2: h (f32) -> bf16 into a_cat[:, 256:384]
// ---------------------------------------------------------------------------
__global__ void k_h_to_acat(const float* __restrict__ h, bf16_t* __restrict__ a_cat) {
  int i = blockIdx.x * blockDim.x + threadIdx.x;   // over M_TOT*D_
  int g = i >> 7, f = i & 127;
  a_cat[(size_t)g * TD + 256 + f] = (bf16_t)h[i];
}

// ---------------------------------------------------------------------------
// K3: ins/outs projection per (e,io): out_t[b][f][n*E+e] = h @ W[e] + bias
// ---------------------------------------------------------------------------
__global__ __launch_bounds__(256)
void k_proj(const bf16_t* __restrict__ a_cat,
            const bf16_t* __restrict__ w_in_t, const bf16_t* __restrict__ w_out_t,
            const float*  __restrict__ b_in,   const float*  __restrict__ b_out,
            bf16_t* __restrict__ ins_t,        bf16_t* __restrict__ outs_t) {
  int lane = threadIdx.x & 31, wave = threadIdx.x >> 5;
  int m0 = blockIdx.x * 32;
  int e  = blockIdx.y & 1;
  int io = blockIdx.y >> 1;
  const bf16_t* Wt   = (io ? w_out_t : w_in_t) + (size_t)e * D_ * D_;
  const float*  bias = (io ? b_out   : b_in)   + e * D_;
  bf16_t*       Ot   = (io ? outs_t  : ins_t);

  int tc = wave;
  v8f acc0 = {0,0,0,0,0,0,0,0};
  v8f acc1 = {0,0,0,0,0,0,0,0};
  const bf16_t* Abase = a_cat + (size_t)m0 * TD + 256;
#pragma unroll
  for (int kk = 0; kk < D_; kk += 32) {
    v16bf bfrag = load_b_t(Wt + (size_t)(tc * 16) * D_ + kk, D_, lane);
    v16bf a0 = load_a_rm(Abase + kk, TD, lane);
    v16bf a1 = load_a_rm(Abase + (size_t)16 * TD + kk, TD, lane);
    acc0 = WMMA_BF16(a0, bfrag, acc0);
    acc1 = WMMA_BF16(a1, bfrag, acc1);
  }
  int f  = tc * 16 + (lane & 15);
  float bv = bias[f];
  int b     = m0 / N_;
  int nbase = m0 - b * N_;
  bf16_t* obase = Ot + (size_t)b * D_ * NE + (size_t)f * NE + e;
  int rofs = (lane >> 4) * 8;
#pragma unroll
  for (int r = 0; r < 8; ++r) {
    int n0 = nbase + rofs + r;
    obase[(size_t)n0 * E_]        = (bf16_t)(acc0[r] + bv);
    obase[(size_t)(n0 + 16) * E_] = (bf16_t)(acc1[r] + bv);
  }
}

// ---------------------------------------------------------------------------
// K4a (TDM path): a_{in,out}[b] = A_{in,out}[b] @ {ins,outs}[b], A pre-converted
// bf16. 16KB A macro-tiles DMAd by the Tensor Data Mover, double-buffered.
// ---------------------------------------------------------------------------
__global__ __launch_bounds__(256)
void k_agg_tdm(const bf16_t* __restrict__ Abf,     // (B, N, 2*NE) bf16
               const bf16_t* __restrict__ ins_t,   // (B, D, NE)
               const bf16_t* __restrict__ outs_t,
               bf16_t* __restrict__ a_cat) {       // (M_TOT, TD)
  __shared__ __align__(16) bf16_t As[2][32][LDP];
  int lane = threadIdx.x & 31, wave = threadIdx.x >> 5;
  int m0 = blockIdx.x * 32;
  int b  = blockIdx.y >> 1;
  int io = blockIdx.y & 1;
  const bf16_t* Ab = Abf + ((size_t)b * N_ + m0) * (2 * NE) + (io ? NE : 0);
  const bf16_t* Bt = (io ? outs_t : ins_t) + (size_t)b * D_ * NE;

  int tc = wave;
  v8f acc0 = {0,0,0,0,0,0,0,0};
  v8f acc1 = {0,0,0,0,0,0,0,0};
  int col = lane & 15, kb = (lane >> 4) * 8;
  const bf16_t* bptr = Bt + (size_t)(tc * 16 + col) * NE + kb;
  int row = lane & 15, klo = (lane >> 4) * 8;

  unsigned lds0 = lds_off(&As[0][0][0]);
  unsigned lds1 = lds_off(&As[1][0][0]);
  if (wave == 0) tdm_load_tile(Ab, lds0);

  constexpr int NT = NE / KT;   // 16 macro-tiles
  for (int t = 0; t < NT; ++t) {
    if (wave == 0) {
      if (t + 1 < NT) {
        tdm_load_tile(Ab + (t + 1) * KT, (t & 1) ? lds0 : lds1);
        __builtin_amdgcn_s_wait_tensorcnt(1);   // tile t complete
      } else {
        __builtin_amdgcn_s_wait_tensorcnt(0);
      }
    }
    __syncthreads();
    const bf16_t (*Asb)[LDP] = As[t & 1];
#pragma unroll
    for (int s = 0; s < KT / 32; ++s) {
      int k0 = s * 32;
      v16bf a0 = cat16(*(const v8bf*)&Asb[row][k0 + klo],
                       *(const v8bf*)&Asb[row][k0 + klo + 16]);
      v16bf a1 = cat16(*(const v8bf*)&Asb[row + 16][k0 + klo],
                       *(const v8bf*)&Asb[row + 16][k0 + klo + 16]);
      int kk = t * KT + k0;
      v16bf bb = cat16(*(const v8bf*)(bptr + kk),
                       *(const v8bf*)(bptr + kk + 16));
      acc0 = WMMA_BF16(a0, bb, acc0);
      acc1 = WMMA_BF16(a1, bb, acc1);
    }
    __syncthreads();
  }
  int f = tc * 16 + (lane & 15);
  size_t gro = ((size_t)b * N_ + m0) * TD + io * D_ + f;
  int rofs = (lane >> 4) * 8;
#pragma unroll
  for (int r = 0; r < 8; ++r) {
    a_cat[gro + (size_t)(rofs + r) * TD]      = (bf16_t)acc0[r];
    a_cat[gro + (size_t)(rofs + r + 16) * TD] = (bf16_t)acc1[r];
  }
}

// ---------------------------------------------------------------------------
// K4b (fallback, small workspace): f32 A converted to bf16 on load via LDS
// ---------------------------------------------------------------------------
__global__ __launch_bounds__(256)
void k_agg(const float* __restrict__ A,
           const bf16_t* __restrict__ ins_t,
           const bf16_t* __restrict__ outs_t,
           bf16_t* __restrict__ a_cat) {
  __shared__ __align__(16) bf16_t As[32][40];
  int lane = threadIdx.x & 31, wave = threadIdx.x >> 5;
  int m0 = blockIdx.x * 32;
  int b  = blockIdx.y >> 1;
  int io = blockIdx.y & 1;
  const float*  Ab = A + ((size_t)b * N_ + m0) * (2 * NE) + (io ? NE : 0);
  const bf16_t* Bt = (io ? outs_t : ins_t) + (size_t)b * D_ * NE;

  int tc = wave;
  v8f acc0 = {0,0,0,0,0,0,0,0};
  v8f acc1 = {0,0,0,0,0,0,0,0};
  int lr = threadIdx.x >> 3;
  int lc = (threadIdx.x & 7) * 4;
  const float* aptr = Ab + (size_t)lr * (2 * NE) + lc;
  int col = lane & 15, kb = (lane >> 4) * 8;
  const bf16_t* bptr = Bt + (size_t)(tc * 16 + col) * NE + kb;
  int row = lane & 15, klo = (lane >> 4) * 8;

  for (int kk = 0; kk < NE; kk += 32) {
    v4f av = *(const v4f*)(aptr + kk);
    if (kk + 32 < NE) __builtin_prefetch(aptr + kk + 32, 0, 1);
    v4bf ac;
    ac[0] = (bf16_t)av[0]; ac[1] = (bf16_t)av[1];
    ac[2] = (bf16_t)av[2]; ac[3] = (bf16_t)av[3];
    *(v4bf*)&As[lr][lc] = ac;
    __syncthreads();
    v16bf a0 = cat16(*(const v8bf*)&As[row][klo],      *(const v8bf*)&As[row][klo + 16]);
    v16bf a1 = cat16(*(const v8bf*)&As[row + 16][klo], *(const v8bf*)&As[row + 16][klo + 16]);
    v16bf bb = cat16(*(const v8bf*)(bptr + kk),        *(const v8bf*)(bptr + kk + 16));
    acc0 = WMMA_BF16(a0, bb, acc0);
    acc1 = WMMA_BF16(a1, bb, acc1);
    __syncthreads();
  }
  int f = tc * 16 + (lane & 15);
  size_t gro = ((size_t)b * N_ + m0) * TD + io * D_ + f;
  int rofs = (lane >> 4) * 8;
#pragma unroll
  for (int r = 0; r < 8; ++r) {
    a_cat[gro + (size_t)(rofs + r) * TD]      = (bf16_t)acc0[r];
    a_cat[gro + (size_t)(rofs + r + 16) * TD] = (bf16_t)acc1[r];
  }
}

// ---------------------------------------------------------------------------
// K6: gates.  which==0: rh = sigmoid(a@W_r+b_r)*h (bf16); which==1: z (f32)
// ---------------------------------------------------------------------------
__global__ __launch_bounds__(256)
void k_gate(const bf16_t* __restrict__ a_cat,
            const bf16_t* __restrict__ w_r_t, const float* __restrict__ b_r,
            const bf16_t* __restrict__ w_z_t, const float* __restrict__ b_z,
            const float*  __restrict__ h,
            bf16_t* __restrict__ rh, float* __restrict__ z) {
  int lane = threadIdx.x & 31, wave = threadIdx.x >> 5;
  int m0 = blockIdx.x * 32;
  int which = blockIdx.y;
  const bf16_t* Wt   = which ? w_z_t : w_r_t;
  const float*  bias = which ? b_z   : b_r;
  int tc = wave;
  v8f acc0 = {0,0,0,0,0,0,0,0};
  v8f acc1 = {0,0,0,0,0,0,0,0};
  const bf16_t* Abase = a_cat + (size_t)m0 * TD;
#pragma unroll
  for (int kk = 0; kk < TD; kk += 32) {
    v16bf bfrag = load_b_t(Wt + (size_t)(tc * 16) * TD + kk, TD, lane);
    v16bf a0 = load_a_rm(Abase + kk, TD, lane);
    v16bf a1 = load_a_rm(Abase + (size_t)16 * TD + kk, TD, lane);
    acc0 = WMMA_BF16(a0, bfrag, acc0);
    acc1 = WMMA_BF16(a1, bfrag, acc1);
  }
  int f  = tc * 16 + (lane & 15);
  float bv = bias[f];
  int rofs = (lane >> 4) * 8;
#pragma unroll
  for (int r = 0; r < 8; ++r) {
    int row0 = m0 + rofs + r;
    int row1 = row0 + 16;
    float v0 = 1.f / (1.f + expf(-(acc0[r] + bv)));
    float v1 = 1.f / (1.f + expf(-(acc1[r] + bv)));
    if (which == 0) {
      rh[(size_t)row0 * D_ + f] = (bf16_t)(v0 * h[(size_t)row0 * D_ + f]);
      rh[(size_t)row1 * D_ + f] = (bf16_t)(v1 * h[(size_t)row1 * D_ + f]);
    } else {
      z[(size_t)row0 * D_ + f] = v0;
      z[(size_t)row1 * D_ + f] = v1;
    }
  }
}

// ---------------------------------------------------------------------------
// K8: h_hat = tanh([a_in|a_out|r*h] @ W_h + b_h); h = (1-z)h + z*h_hat
// ---------------------------------------------------------------------------
__global__ __launch_bounds__(256)
void k_update(const bf16_t* __restrict__ a_cat,
              const bf16_t* __restrict__ rh,
              const bf16_t* __restrict__ w_h_t, const float* __restrict__ b_h,
              const float*  __restrict__ z,
              float* __restrict__ h) {
  int lane = threadIdx.x & 31, wave = threadIdx.x >> 5;
  int m0 = blockIdx.x * 32;
  int tc = wave;
  v8f acc0 = {0,0,0,0,0,0,0,0};
  v8f acc1 = {0,0,0,0,0,0,0,0};
#pragma unroll
  for (int kk = 0; kk < TD; kk += 32) {
    v16bf bfrag = load_b_t(w_h_t + (size_t)(tc * 16) * TD + kk, TD, lane);
    v16bf a0, a1;
    if (kk < 2 * D_) {
      a0 = load_a_rm(a_cat + (size_t)m0 * TD + kk, TD, lane);
      a1 = load_a_rm(a_cat + (size_t)(m0 + 16) * TD + kk, TD, lane);
    } else {
      a0 = load_a_rm(rh + (size_t)m0 * D_ + (kk - 2 * D_), D_, lane);
      a1 = load_a_rm(rh + (size_t)(m0 + 16) * D_ + (kk - 2 * D_), D_, lane);
    }
    acc0 = WMMA_BF16(a0, bfrag, acc0);
    acc1 = WMMA_BF16(a1, bfrag, acc1);
  }
  int f  = tc * 16 + (lane & 15);
  float bv = b_h[f];
  int rofs = (lane >> 4) * 8;
#pragma unroll
  for (int r = 0; r < 8; ++r) {
    int row0 = m0 + rofs + r;
    int row1 = row0 + 16;
    size_t i0 = (size_t)row0 * D_ + f;
    size_t i1 = (size_t)row1 * D_ + f;
    float t0 = tanhf(acc0[r] + bv);
    float t1 = tanhf(acc1[r] + bv);
    float z0 = z[i0], z1 = z[i1];
    h[i0] = (1.f - z0) * h[i0] + z0 * t0;
    h[i1] = (1.f - z1) * h[i1] + z1 * t1;
  }
}

// ---------------------------------------------------------------------------
extern "C" void kernel_launch(void* const* d_in, const int* in_sizes, int n_in,
                              void* d_out, int out_size, void* d_ws, size_t ws_size,
                              hipStream_t stream) {
  (void)in_sizes; (void)n_in; (void)out_size;
  const float* prop  = (const float*)d_in[0];
  const float* A     = (const float*)d_in[1];
  const float* W_in  = (const float*)d_in[2];
  const float* b_in  = (const float*)d_in[3];
  const float* W_out = (const float*)d_in[4];
  const float* b_out = (const float*)d_in[5];
  const float* W_r   = (const float*)d_in[6];
  const float* b_r   = (const float*)d_in[7];
  const float* W_z   = (const float*)d_in[8];
  const float* b_z   = (const float*)d_in[9];
  const float* W_h   = (const float*)d_in[10];
  const float* b_h   = (const float*)d_in[11];
  float* h = (float*)d_out;

  char* wsp = (char*)d_ws;
  auto alloc = [&](size_t bytes) {
    char* p = wsp;
    wsp += (bytes + 255) & ~(size_t)255;
    return p;
  };
  bf16_t* a_cat   = (bf16_t*)alloc((size_t)M_TOT * TD * 2);
  bf16_t* ins_t   = (bf16_t*)alloc((size_t)B_ * D_ * NE * 2);
  bf16_t* outs_t  = (bf16_t*)alloc((size_t)B_ * D_ * NE * 2);
  bf16_t* rh      = (bf16_t*)alloc((size_t)M_TOT * D_ * 2);
  float*  zbuf    = (float*) alloc((size_t)M_TOT * D_ * 4);
  bf16_t* w_in_t  = (bf16_t*)alloc((size_t)E_ * D_ * D_ * 2);
  bf16_t* w_out_t = (bf16_t*)alloc((size_t)E_ * D_ * D_ * 2);
  bf16_t* w_r_t   = (bf16_t*)alloc((size_t)TD * D_ * 2);
  bf16_t* w_z_t   = (bf16_t*)alloc((size_t)TD * D_ * 2);
  bf16_t* w_h_t   = (bf16_t*)alloc((size_t)TD * D_ * 2);
  bf16_t* A_bf    = (bf16_t*)alloc((size_t)B_ * N_ * 2 * NE * 2);   // 128MB, last
  size_t need = (size_t)(wsp - (char*)d_ws);
  bool big = (need <= ws_size);

  hipMemcpyAsync(h, prop, (size_t)M_TOT * D_ * sizeof(float),
                 hipMemcpyDeviceToDevice, stream);

  const int TPB = 256;
  int gw = (D_ * D_ + TPB - 1) / TPB;
  k_transpose_bf16<<<gw, TPB, 0, stream>>>(W_in,            w_in_t,           D_, D_);
  k_transpose_bf16<<<gw, TPB, 0, stream>>>(W_in + D_ * D_,  w_in_t + D_ * D_, D_, D_);
  k_transpose_bf16<<<gw, TPB, 0, stream>>>(W_out,           w_out_t,          D_, D_);
  k_transpose_bf16<<<gw, TPB, 0, stream>>>(W_out + D_ * D_, w_out_t + D_ * D_,D_, D_);
  int gw3 = (TD * D_ + TPB - 1) / TPB;
  k_transpose_bf16<<<gw3, TPB, 0, stream>>>(W_r, w_r_t, TD, D_);
  k_transpose_bf16<<<gw3, TPB, 0, stream>>>(W_z, w_z_t, TD, D_);
  k_transpose_bf16<<<gw3, TPB, 0, stream>>>(W_h, w_h_t, TD, D_);

  if (big) {
    size_t nA = (size_t)B_ * N_ * 2 * NE;   // 64M elems
    k_conv_A<<<(unsigned)(nA / (8 * TPB)), TPB, 0, stream>>>(A, A_bf);
  }

  for (int step = 0; step < NSTEP; ++step) {
    k_h_to_acat<<<(M_TOT * D_) / TPB, TPB, 0, stream>>>(h, a_cat);
    k_proj<<<dim3(M_TOT / 32, 4), TPB, 0, stream>>>(a_cat, w_in_t, w_out_t,
                                                    b_in, b_out, ins_t, outs_t);
    if (big)
      k_agg_tdm<<<dim3(N_ / 32, B_ * 2), TPB, 0, stream>>>(A_bf, ins_t, outs_t, a_cat);
    else
      k_agg<<<dim3(N_ / 32, B_ * 2), TPB, 0, stream>>>(A, ins_t, outs_t, a_cat);
    k_gate<<<dim3(M_TOT / 32, 2), TPB, 0, stream>>>(a_cat, w_r_t, b_r,
                                                    w_z_t, b_z, h, rh, zbuf);
    k_update<<<dim3(M_TOT / 32, 1), TPB, 0, stream>>>(a_cat, rh, w_h_t, b_h, zbuf, h);
  }
}
